// ALiBiAttention_61237643706932
// MI455X (gfx1250) — matrix operand order repaired and checked by reading, not compiled, not run on hardware
//
#include <hip/hip_runtime.h>

#define DM    1024
#define NH    16
#define HD    64
#define SEQ   2048
#define BATCH 2
#define MROWS (BATCH*SEQ)   // 4096 rows of the flattened activation matrix

typedef __attribute__((ext_vector_type(16))) __bf16         v16bf;
typedef __attribute__((ext_vector_type(8)))  float          v8f;
typedef __attribute__((ext_vector_type(8)))  unsigned short v8us;

// Two 16B halves viewed as one WMMA 16-element bf16 fragment.
union BFrag {
    struct { v8us lo, hi; } u;
    v16bf v;
};

__device__ __forceinline__ unsigned short f2bf(float f) {
    union { float f; unsigned int u; } c; c.f = f;
    unsigned int r = c.u + 0x7FFFu + ((c.u >> 16) & 1u);   // round-to-nearest-even
    return (unsigned short)(r >> 16);
}

__device__ __forceinline__ v8f wmma_bf16(v16bf a, v16bf b, v8f c) {
    // D(f32 16x16) = A(bf16 16x32) * B(bf16 32x16) + C
    return __builtin_amdgcn_wmma_f32_16x16x32_bf16(false, a, false, b, (short)0, c, false, false);
}

__device__ __forceinline__ v8f vzero8() {
    v8f z;
#pragma unroll
    for (int i = 0; i < 8; ++i) z[i] = 0.0f;
    return z;
}

// VALU butterfly max within 16-lane rows via DPP ROW_XMASK (0x160 | xor_mask).
template<int CTRL>
__device__ __forceinline__ float dpp_max_step(float x) {
    int y = __builtin_amdgcn_update_dpp(0, __builtin_bit_cast(int, x),
                                        CTRL, 0xF, 0xF, true);
    return fmaxf(x, __builtin_bit_cast(float, y));
}
__device__ __forceinline__ float row16_max(float x) {
    x = dpp_max_step<0x161>(x);   // xor 1
    x = dpp_max_step<0x162>(x);   // xor 2
    x = dpp_max_step<0x164>(x);   // xor 4
    x = dpp_max_step<0x168>(x);   // xor 8
    return x;
}

// ---------------------------------------------------------------- conversion
__global__ void cvt_f32_bf16(const float* __restrict__ in,
                             unsigned short* __restrict__ out, int n) {
    int i = blockIdx.x * blockDim.x + threadIdx.x;
    if (i < n) out[i] = f2bf(in[i]);
}

// ---------------------------------------------------------------- GEMM
// y[M,N] = scale * (A[M,K](bf16) * W[N,K](bf16, torch layout => implicit W^T) + bias[N])
// MODE 0: store bf16 into [B,H,S,hd]        (Q with scale=1/8, K with scale=1)
// MODE 1: store bf16 into [B,H,hd,S]        (V transposed for attention B-frags)
// MODE 2: store f32 row-major [M,N]         (final output)
// Each wave computes a 32x64 tile; block = 8 waves => 256x64 block tile.
template<int MODE>
__global__ void gemm_wmma(const unsigned short* __restrict__ A,
                          const unsigned short* __restrict__ W,
                          const float* __restrict__ bias,
                          void* __restrict__ out, float scale) {
    const int K   = DM, N = DM;
    const int lane = threadIdx.x & 31;
    const int wv   = threadIdx.x >> 5;
    const int ln   = lane & 15;
    const int hi   = lane >> 4;                 // lane-half select
    const int m0   = (blockIdx.x * 8 + wv) * 32;
    const int n0   = blockIdx.y * 64;

    v8f acc[2][4];
#pragma unroll
    for (int mt = 0; mt < 2; ++mt)
#pragma unroll
        for (int nt = 0; nt < 4; ++nt) acc[mt][nt] = vzero8();

    const unsigned short* arow[2] = { A + (size_t)(m0 + ln) * K,
                                      A + (size_t)(m0 + 16 + ln) * K };

    // unroll bounded to 4: enough to fill s_clause batches and cover WMMA
    // latency without blowing up the WGP I$ (full unroll was 256 WMMAs/kernel)
#pragma unroll 4
    for (int k0 = 0; k0 < K; k0 += 32) {
        // A-frags: lanes 0-15 K {0..15}, lanes 16-31 K {16..31}
        BFrag af[2];
#pragma unroll
        for (int mt = 0; mt < 2; ++mt) {
            af[mt].u.lo = *(const v8us*)(arow[mt] + k0 + hi * 16);
            af[mt].u.hi = *(const v8us*)(arow[mt] + k0 + 8 + hi * 16);
        }
#pragma unroll
        for (int nt = 0; nt < 4; ++nt) {
            // B-frag: column n = weight row n; 16 contiguous K values per lane
            const unsigned short* wp = W + (size_t)(n0 + nt * 16 + ln) * K + k0 + hi * 16;
            BFrag bf_;
            bf_.u.lo = *(const v8us*)(wp);
            bf_.u.hi = *(const v8us*)(wp + 8);
#pragma unroll
            for (int mt = 0; mt < 2; ++mt)
                acc[mt][nt] = wmma_bf16(af[mt].v, bf_.v, acc[mt][nt]);
        }
    }

#pragma unroll
    for (int nt = 0; nt < 4; ++nt) {
        const int n  = n0 + nt * 16 + ln;
        const float bv = bias[n];
#pragma unroll
        for (int mt = 0; mt < 2; ++mt) {
#pragma unroll
            for (int r = 0; r < 8; ++r) {
                const int m = m0 + mt * 16 + r + hi * 8;  // C layout row
                float vval = (acc[mt][nt][r] + bv) * scale;
                if (MODE == 2) {
                    ((float*)out)[(size_t)m * N + n] = vval;
                } else {
                    const int b = m >> 11, s = m & (SEQ - 1);
                    const int h = n >> 6,  d = n & (HD - 1);
                    size_t idx;
                    if (MODE == 0) idx = (((size_t)(b * NH + h) * SEQ + s) * HD + d);
                    else           idx = (((size_t)(b * NH + h) * HD + d) * SEQ + s);
                    ((unsigned short*)out)[idx] = f2bf(vval);
                }
            }
        }
    }
}

// ---------------------------------------------------------------- attention
// One wave per 16-query tile; streams 32-key blocks with online softmax.
// Q is pre-scaled by 1/sqrt(hd) in its projection GEMM.
__global__ void attn_wmma(const unsigned short* __restrict__ Q,   // [B,H,S,hd]
                          const unsigned short* __restrict__ Km,  // [B,H,S,hd]
                          const unsigned short* __restrict__ Vt,  // [B,H,hd,S]
                          unsigned short* __restrict__ O) {       // [B,S,D] bf16
    __shared__ __align__(16) unsigned short pbuf[8][16 * 32];     // per-wave P staging

    const int lane = threadIdx.x & 31;
    const int wv   = threadIdx.x >> 5;
    const int ln   = lane & 15;
    const int hi   = lane >> 4;
    const int wid  = blockIdx.x * 8 + wv;
    const int qt   = wid & (SEQ / 16 - 1);
    const int bh   = wid >> 7;                  // b*NH + h
    const int h    = bh & (NH - 1);
    const int q0   = qt * 16;
    const float slope = exp2f(-0.5f * (float)(h + 1));   // ALiBi, H=16 power of two

    const unsigned short* Qh = Q  + (size_t)bh * SEQ * HD;
    const unsigned short* Kh = Km + (size_t)bh * SEQ * HD;
    const unsigned short* Vh = Vt + (size_t)bh * HD * SEQ;

    // Q A-fragments for both 32-wide K-dim chunks of hd=64
    BFrag qa[2];
    {
        const unsigned short* qrow = Qh + (size_t)(q0 + ln) * HD;
#pragma unroll
        for (int f = 0; f < 2; ++f) {
            qa[f].u.lo = *(const v8us*)(qrow + f * 32 + hi * 16);
            qa[f].u.hi = *(const v8us*)(qrow + f * 32 + 8 + hi * 16);
        }
    }

    // all-ones B fragment: one WMMA against it yields per-row sums of P,
    // replicated across the row's lanes in C layout (replaces shuffle-add tree)
    BFrag ones;
#pragma unroll
    for (int i = 0; i < 8; ++i) { ones.u.lo[i] = 0x3F80; ones.u.hi[i] = 0x3F80; }

    v8f o[4];
#pragma unroll
    for (int nt = 0; nt < 4; ++nt) o[nt] = vzero8();
    float mrow[8], lrow[8], qbias[8];
#pragma unroll
    for (int r = 0; r < 8; ++r) {
        mrow[r]  = -3.0e38f;
        lrow[r]  = 0.0f;
        qbias[r] = slope * (float)(q0 + r + hi * 8);      // loop-invariant ALiBi term
    }

    unsigned short* pb = pbuf[wv];

    for (int j0 = 0; j0 < q0 + 16; j0 += 32) {   // causal: only blocks touching <= q
        // ---- S = Q K^T for two 16-key sub-tiles
        v8f sc[2];
#pragma unroll
        for (int jj = 0; jj < 2; ++jj) {
            v8f s = vzero8();
#pragma unroll
            for (int f = 0; f < 2; ++f) {
                const unsigned short* kp =
                    Kh + (size_t)(j0 + jj * 16 + ln) * HD + f * 32 + hi * 16;
                BFrag kf;
                kf.u.lo = *(const v8us*)(kp);
                kf.u.hi = *(const v8us*)(kp + 8);
                s = wmma_bf16(qa[f].v, kf.v, s);
            }
            sc[jj] = s;
        }

        // ---- ALiBi + causal mask + online softmax (C layout; DPP row max)
        const float cb0 = slope * (float)(j0 + ln);
        const float cb1 = slope * (float)(j0 + 16 + ln);
        const int   c0  = j0 + ln, c1 = j0 + 16 + ln;
        float alf[8];
#pragma unroll
        for (int r = 0; r < 8; ++r) {
            const int qr = q0 + r + hi * 8;
            float s0 = sc[0][r] + (cb0 - qbias[r]);
            float s1 = sc[1][r] + (cb1 - qbias[r]);
            if (c0 > qr) s0 = -3.0e38f;
            if (c1 > qr) s1 = -3.0e38f;

            const float mr = row16_max(fmaxf(s0, s1));
            const float mn = fmaxf(mrow[r], mr);
            alf[r] = __expf(mrow[r] - mn);
            mrow[r] = mn;
            const float p0 = __expf(s0 - mn);
            const float p1 = __expf(s1 - mn);

            const int prow = r + hi * 8;        // C layout -> LDS (row-major 16x32)
            pb[prow * 32 + ln]      = f2bf(p0);
            pb[prow * 32 + 16 + ln] = f2bf(p1);
        }
#pragma unroll
        for (int nt = 0; nt < 4; ++nt)
#pragma unroll
            for (int r = 0; r < 8; ++r) o[nt][r] *= alf[r];

        // ---- P (C layout) -> A-fragment layout via LDS (same-wave, DS-ordered)
        BFrag pa;
        pa.u.lo = *(const v8us*)(pb + ln * 32 + hi * 16);
        pa.u.hi = *(const v8us*)(pb + ln * 32 + 8 + hi * 16);

        // ---- row sums of P via WMMA against all-ones B
        const v8f rs = wmma_bf16(pa.v, ones.v, vzero8());
#pragma unroll
        for (int r = 0; r < 8; ++r) lrow[r] = lrow[r] * alf[r] + rs[r];

        // ---- O += P * V  (V^T layout makes B-frags contiguous per lane)
#pragma unroll
        for (int nt = 0; nt < 4; ++nt) {
            const unsigned short* vp = Vh + (size_t)(nt * 16 + ln) * SEQ + j0 + hi * 16;
            BFrag vf;
            vf.u.lo = *(const v8us*)(vp);
            vf.u.hi = *(const v8us*)(vp + 8);
            o[nt] = wmma_bf16(pa.v, vf.v, o[nt]);
        }
    }

    // ---- normalize and store attn output bf16 in [B,S,H*hd] (= GEMM A layout)
    const int b = bh >> 4;
#pragma unroll
    for (int r = 0; r < 8; ++r) {
        const float inv = 1.0f / lrow[r];
        const int srow = q0 + r + hi * 8;
#pragma unroll
        for (int nt = 0; nt < 4; ++nt) {
            const int col = h * HD + nt * 16 + ln;
            O[((size_t)b * SEQ + srow) * DM + col] = f2bf(o[nt][r] * inv);
        }
    }
}

// ---------------------------------------------------------------- launch
extern "C" void kernel_launch(void* const* d_in, const int* in_sizes, int n_in,
                              void* d_out, int out_size, void* d_ws, size_t ws_size,
                              hipStream_t stream) {
    const float* x  = (const float*)d_in[0];
    const float* qw = (const float*)d_in[1];
    const float* qb = (const float*)d_in[2];
    const float* kw = (const float*)d_in[3];
    const float* kb = (const float*)d_in[4];
    const float* vw = (const float*)d_in[5];
    const float* vb = (const float*)d_in[6];
    const float* ow = (const float*)d_in[7];
    const float* ob = (const float*)d_in[8];

    char* ws = (char*)d_ws;
    const size_t actB = (size_t)MROWS * DM * sizeof(unsigned short);  // 8 MB
    const size_t wB   = (size_t)DM * DM * sizeof(unsigned short);     // 2 MB
    unsigned short* xb  = (unsigned short*)ws;            ws += actB;
    unsigned short* wqb = (unsigned short*)ws;            ws += wB;
    unsigned short* wkb = (unsigned short*)ws;            ws += wB;
    unsigned short* wvb = (unsigned short*)ws;            ws += wB;
    unsigned short* wob = (unsigned short*)ws;            ws += wB;
    unsigned short* Qb  = (unsigned short*)ws;            ws += actB;
    unsigned short* Kb  = (unsigned short*)ws;            ws += actB;
    unsigned short* Vt  = (unsigned short*)ws;            ws += actB;
    unsigned short* Ab  = (unsigned short*)ws;            ws += actB;

    const int nAct = MROWS * DM, nW = DM * DM;
    cvt_f32_bf16<<<(nAct + 255) / 256, 256, 0, stream>>>(x,  xb,  nAct);
    cvt_f32_bf16<<<(nW   + 255) / 256, 256, 0, stream>>>(qw, wqb, nW);
    cvt_f32_bf16<<<(nW   + 255) / 256, 256, 0, stream>>>(kw, wkb, nW);
    cvt_f32_bf16<<<(nW   + 255) / 256, 256, 0, stream>>>(vw, wvb, nW);
    cvt_f32_bf16<<<(nW   + 255) / 256, 256, 0, stream>>>(ow, wob, nW);

    dim3 gg(MROWS / 256, DM / 64);               // (16, 16) blocks of 8 waves
    gemm_wmma<0><<<gg, 256, 0, stream>>>(xb, wqb, qb, Qb, 0.125f);  // fold 1/sqrt(hd)
    gemm_wmma<0><<<gg, 256, 0, stream>>>(xb, wkb, kb, Kb, 1.0f);
    gemm_wmma<1><<<gg, 256, 0, stream>>>(xb, wvb, vb, Vt, 1.0f);

    attn_wmma<<<(BATCH * NH * (SEQ / 16)) / 8, 256, 0, stream>>>(Qb, Kb, Vt, Ab);

    gemm_wmma<2><<<gg, 256, 0, stream>>>(Ab, wob, ob, d_out, 1.0f);
}
